// BCNpp_46892452938168
// MI455X (gfx1250) — matrix operand order repaired and checked by reading, not compiled, not run on hardware
//
#include <hip/hip_runtime.h>
#include <hip/hip_bf16.h>

typedef __attribute__((ext_vector_type(16))) _Float16 v16h;
typedef __attribute__((ext_vector_type(8)))  _Float16 v8h;
typedef __attribute__((ext_vector_type(8)))  float    v8f;

#define THETA 0.7f

// =====================================================================
// Generic 3x3 conv (CDC pre-folded into weights) + BN + ReLU.
// Implicit GEMM with V_WMMA_F32_16X16X32_F16.
//   activations: NHWC f16, (H+2)x(W+2) with zero halo, channels padded to
//                Cinp (multiple of 32)
//   weights:     f16 [co][k3][ci_pad], CDC-folded, zero-padded channels
// Each wave computes a 16(cout) x 16(pixel) tile. H*W and Cout are always
// multiples of 16 here, so no masking is needed (EXEC all-ones for WMMA).
// Fragment layouts per CDNA5 ISA 7.12.2:
//   A (16x32 f16): lane l holds row M=l&15; K runs [hi*8, hi*8+7] and
//                  [16+hi*8, 16+hi*8+7]  (hi = l>>4)
//   B (32x16 f16): lane l holds col N=l&15; K runs [hi*8, hi*8+7] and [16+hi*8, 16+hi*8+7]
//   D (16x16 f32): VGPR r, lane l -> M = r + hi*8, N = l&15
// =====================================================================
__global__ __launch_bounds__(256) void conv3_wmma(
    const _Float16* __restrict__ in, const _Float16* __restrict__ wt,
    const float* __restrict__ sc, const float* __restrict__ bi,
    _Float16* __restrict__ out,
    int H, int W, int Cinp, int Cout, int Coutp, int ntiles)
{
  int wave = blockIdx.x * 8 + (threadIdx.x >> 5);
  if (wave >= ntiles) return;
  int lane = threadIdx.x & 31;
  int nCo = Cout >> 4;
  int cot = wave % nCo;       // adjacent waves share the same pixel tile
  int pt  = wave / nCo;
  int lp  = lane & 15;
  int hi  = lane >> 4;
  int p = pt * 16 + lp;
  int y = p / W, x = p % W;
  int co = cot * 16 + lp;
  int Wp = W + 2;
  const _Float16* wrow = wt + (size_t)co * 9 * Cinp;
  v8f acc = {};
  for (int k3 = 0; k3 < 9; ++k3) {
    int dy = k3 / 3 - 1, dx = k3 % 3 - 1;
    const _Float16* asrc = wrow + k3 * Cinp;
    const _Float16* bsrc = in + ((size_t)(y + 1 + dy) * Wp + (x + 1 + dx)) * Cinp;
    for (int cb = 0; cb < Cinp; cb += 32) {
      v8h alo = *(const v8h*)(asrc + cb + hi * 8);
      v8h ahi = *(const v8h*)(asrc + cb + 16 + hi * 8);
      v16h a = __builtin_shufflevector(alo, ahi,
                 0,1,2,3,4,5,6,7,8,9,10,11,12,13,14,15);
      v8h blo = *(const v8h*)(bsrc + cb + hi * 8);
      v8h bhi = *(const v8h*)(bsrc + cb + 16 + hi * 8);
      v16h b = __builtin_shufflevector(blo, bhi,
                 0,1,2,3,4,5,6,7,8,9,10,11,12,13,14,15);
      acc = __builtin_amdgcn_wmma_f32_16x16x32_f16(
              false, a, false, b, (short)0, acc, false, false);
    }
  }
  int cbase = cot * 16 + hi * 8;
  v8h o;
  #pragma unroll
  for (int r = 0; r < 8; ++r) {
    float v = acc[r] * sc[cbase + r] + bi[cbase + r];
    o[r] = (_Float16)(v > 0.f ? v : 0.f);
  }
  *(v8h*)(out + ((size_t)(y + 1) * Wp + (x + 1)) * Coutp + cbase) = o;
}

// ---------------------------------------------------------------------
// Weight transform: w (Cout,Cin,3,3) f32 -> f16 [co][k3][ci_pad], with the
// CDC 1x1 kernel-sum folded into the center tap. Pad channels stay zero
// (workspace is memset before this runs).
// ---------------------------------------------------------------------
__global__ void wtrans(const float* __restrict__ w, _Float16* __restrict__ wt,
                       int Cout, int Cin, int Cinp)
{
  int idx = blockIdx.x * blockDim.x + threadIdx.x;
  if (idx >= Cout * Cin) return;
  int ci = idx % Cin, co = idx / Cin;
  const float* src = w + ((size_t)co * Cin + ci) * 9;
  float s = 0.f;
  #pragma unroll
  for (int k = 0; k < 9; ++k) s += src[k];
  #pragma unroll
  for (int k = 0; k < 9; ++k) {
    float v = src[k] - (k == 4 ? THETA * s : 0.f);
    wt[((size_t)co * 9 + k) * Cinp + ci] = (_Float16)v;
  }
}

// NCHW f32 (3,128,128) -> NHWC f16 halo buffer (130x130x32)
__global__ void pack_x(const float* __restrict__ x, _Float16* __restrict__ out)
{
  int idx = blockIdx.x * blockDim.x + threadIdx.x;
  if (idx >= 3 * 128 * 128) return;
  int c = idx / 16384, p = idx % 16384;
  int y = p / 128, xx = p % 128;
  out[((size_t)(y + 1) * 130 + (xx + 1)) * 32 + c] = (_Float16)x[idx];
}

// DBO: per pixel, pairwise-gaussian channel smoothing (sigma=1).
// One block per pixel; channels staged in LDS.
__global__ void dbo_kernel(const _Float16* __restrict__ in, _Float16* __restrict__ out,
                           int W, int C, int Cp)
{
  __shared__ float xs[256];
  int p = blockIdx.x;
  int y = p / W, x = p % W;
  size_t base = ((size_t)(y + 1) * (W + 2) + (x + 1)) * Cp;
  int t = threadIdx.x;
  if (t < C) xs[t] = (float)in[base + t];
  __syncthreads();
  if (t < C) {
    float xj = xs[t], num = 0.f, den = 0.f;
    for (int i = 0; i < C; ++i) {
      float d = xs[i] - xj;
      float g = __expf(-d * d);
      num += g * xs[i];
      den += g;
    }
    out[base + t] = (_Float16)(num / den);
  }
}

// MaxPool 3x3 stride2 pad1. Inputs are >=0 (post-ReLU sums), so the zero
// halo can never beat an interior value -> matches -inf padding.
__global__ void maxpool_kernel(const _Float16* __restrict__ in, _Float16* __restrict__ out,
                               int Hi, int C, int Cpi, int Cpo)
{
  int Ho = Hi >> 1;
  int idx = blockIdx.x * blockDim.x + threadIdx.x;
  int total = Ho * Ho * C;
  if (idx >= total) return;
  int c = idx % C, p = idx / C;
  int x = p % Ho, y = p / Ho;
  float m = -1e30f;
  #pragma unroll
  for (int ky = 0; ky < 3; ++ky)
    #pragma unroll
    for (int kx = 0; kx < 3; ++kx) {
      int iy = 2 * y - 1 + ky, ix = 2 * x - 1 + kx;
      float v = (float)in[((size_t)(iy + 1) * (Hi + 2) + (ix + 1)) * Cpi + c];
      m = fmaxf(m, v);
    }
  out[((size_t)(y + 1) * (Ho + 2) + (x + 1)) * Cpo + c] = (_Float16)m;
}

__global__ void add_kernel(const _Float16* __restrict__ a, const _Float16* __restrict__ b,
                           _Float16* __restrict__ o, int n)
{
  int i = blockIdx.x * blockDim.x + threadIdx.x;
  if (i < n) o[i] = (_Float16)((float)a[i] + (float)b[i]);
}

// SA step 1: channel mean & max planes (f32)
__global__ void sa_reduce(const _Float16* __restrict__ in, float* __restrict__ avg,
                          float* __restrict__ mx, int H, int C, int Cp)
{
  int p = blockIdx.x * blockDim.x + threadIdx.x;
  if (p >= H * H) return;
  int y = p / H, x = p % H;
  const _Float16* src = in + ((size_t)(y + 1) * (H + 2) + (x + 1)) * Cp;
  float s = 0.f, m = -1e30f;
  for (int c = 0; c < C; ++c) {
    float v = (float)src[c];
    s += v;
    m = fmaxf(m, v);
  }
  avg[p] = s / (float)C;
  mx[p] = m;
}

// SA step 2: kxk conv over (avg,max), zero pad, sigmoid
__global__ void sa_conv(const float* __restrict__ avg, const float* __restrict__ mx,
                        const float* __restrict__ w, int H, int k, float* __restrict__ att)
{
  int p = blockIdx.x * blockDim.x + threadIdx.x;
  if (p >= H * H) return;
  int y = p / H, x = p % H, r = k / 2;
  float acc = 0.f;
  for (int ky = 0; ky < k; ++ky)
    for (int kx = 0; kx < k; ++kx) {
      int iy = y + ky - r, ix = x + kx - r;
      if (iy < 0 || iy >= H || ix < 0 || ix >= H) continue;
      acc += w[ky * k + kx] * avg[iy * H + ix]
           + w[k * k + ky * k + kx] * mx[iy * H + ix];
    }
  att[p] = 1.f / (1.f + __expf(-acc));
}

// att*blk, then jax antialiased 2x downsample ([1,3,3,1]/8, edge renorm)
__global__ void attn_down2(const _Float16* __restrict__ blk, const float* __restrict__ att,
                           _Float16* __restrict__ F, int Hi, int C, int Cp, int coff)
{
  int idx = blockIdx.x * blockDim.x + threadIdx.x;
  int total = 32 * 32 * C;
  if (idx >= total) return;
  int c = idx % C, p = idx / C;
  int x = p % 32, y = p / 32;
  const float wk[4] = {0.125f, 0.375f, 0.375f, 0.125f};
  float acc = 0.f, ws = 0.f;
  for (int ty = 0; ty < 4; ++ty) {
    int iy = 2 * y - 1 + ty;
    if (iy < 0 || iy >= Hi) continue;
    for (int tx = 0; tx < 4; ++tx) {
      int ix = 2 * x - 1 + tx;
      if (ix < 0 || ix >= Hi) continue;
      float w2 = wk[ty] * wk[tx];
      float v = att[iy * Hi + ix] *
                (float)blk[((size_t)(iy + 1) * (Hi + 2) + (ix + 1)) * Cp + c];
      acc += w2 * v;
      ws += w2;
    }
  }
  F[((size_t)(y + 1) * 34 + (x + 1)) * 480 + coff + c] = (_Float16)(acc / ws);
}

// att*blk at 32x32 (resize is identity)
__global__ void attn_ident(const _Float16* __restrict__ blk, const float* __restrict__ att,
                           _Float16* __restrict__ F, int C, int Cp, int coff)
{
  int idx = blockIdx.x * blockDim.x + threadIdx.x;
  int total = 32 * 32 * C;
  if (idx >= total) return;
  int c = idx % C, p = idx / C;
  int x = p % 32, y = p / 32;
  float v = att[p] * (float)blk[((size_t)(y + 1) * 34 + (x + 1)) * Cp + c];
  F[((size_t)(y + 1) * 34 + (x + 1)) * 480 + coff + c] = (_Float16)v;
}

// att*blk, half-pixel bilinear 2x upsample (16 -> 32), edge renorm == clamp
__global__ void attn_up2(const _Float16* __restrict__ blk, const float* __restrict__ att,
                         _Float16* __restrict__ F, int Hi, int C, int Cp, int coff)
{
  int idx = blockIdx.x * blockDim.x + threadIdx.x;
  int total = 32 * 32 * C;
  if (idx >= total) return;
  int c = idx % C, p = idx / C;
  int x = p % 32, y = p / 32;
  float sy = y * 0.5f - 0.25f, sx = x * 0.5f - 0.25f;
  int iy0 = (int)floorf(sy), ix0 = (int)floorf(sx);
  float fy = sy - (float)iy0, fx = sx - (float)ix0;
  float acc = 0.f, ws = 0.f;
  for (int ty = 0; ty < 2; ++ty) {
    int iy = iy0 + ty;
    float wy = ty ? fy : 1.f - fy;
    if (iy < 0 || iy >= Hi) continue;
    for (int tx = 0; tx < 2; ++tx) {
      int ix = ix0 + tx;
      float wx = tx ? fx : 1.f - fx;
      if (ix < 0 || ix >= Hi) continue;
      float v = att[iy * Hi + ix] *
                (float)blk[((size_t)(iy + 1) * (Hi + 2) + (ix + 1)) * Cp + c];
      acc += wy * wx * v;
      ws += wy * wx;
    }
  }
  F[((size_t)(y + 1) * 34 + (x + 1)) * 480 + coff + c] = (_Float16)(acc / ws);
}

// Final cdc 160->1 conv + relu, f32 output. Tiny; folds theta on the fly.
__global__ void last2_kernel(const _Float16* __restrict__ G, const float* __restrict__ w,
                             float* __restrict__ out)
{
  int p = blockIdx.x * blockDim.x + threadIdx.x;
  if (p >= 1024) return;
  int y = p / 32, x = p % 32;
  float acc = 0.f;
  for (int c = 0; c < 160; ++c) {
    const float* wc = w + c * 9;
    float center = (float)G[((size_t)(y + 1) * 34 + (x + 1)) * 160 + c];
    float ws = 0.f;
    #pragma unroll
    for (int k3 = 0; k3 < 9; ++k3) {
      int dy = k3 / 3 - 1, dx = k3 % 3 - 1;
      float v = (float)G[((size_t)(y + 1 + dy) * 34 + (x + 1 + dx)) * 160 + c];
      acc += wc[k3] * v;
      ws += wc[k3];
    }
    acc -= THETA * ws * center;
  }
  out[p] = fmaxf(acc, 0.f);
}

// =====================================================================
extern "C" void kernel_launch(void* const* d_in, const int* in_sizes, int n_in,
                              void* d_out, int out_size, void* d_ws, size_t ws_size,
                              hipStream_t stream)
{
  if (n_in < 71) return;
  // --- locate x (unique 49152-element tensor) and determine param order ---
  int xidx = -1;
  for (int i = 0; i < n_in; ++i)
    if (in_sizes[i] == 3 * 128 * 128) { xidx = i; break; }
  if (xidx < 0) xidx = 0;
  int pbase = (xidx == 0) ? 1 : 0;
  // sorted-key flatten puts b1's (160,80,3,3) weight first (115200);
  // insertion order puts conv1's (80,3,3,3) weight first (2160)
  bool sortedOrder = (in_sizes[pbase] != 80 * 3 * 9);
  enum { G_B1, G_B1D, G_B2, G_B2D, G_B3, G_B3D, G_CONV1, G_LAST1, G_LAST2,
         G_SA1, G_SA2, G_SA3 };
  static const int posS[12] = {0, 9, 18, 30, 42, 51, 60, 63, 66, 67, 68, 69};
  static const int posI[12] = {3, 12, 21, 33, 45, 54, 0, 63, 66, 67, 68, 69};
  const int* pos = sortedOrder ? posS : posI;
  auto Pf = [&](int g, int leaf) -> const float* {
    return (const float*)d_in[pbase + pos[g] + leaf];
  };
  const float* x = (const float*)d_in[xidx];

  // --- workspace bump allocator (256B aligned) ---
  char* base = (char*)d_ws;
  size_t off = 0;
  auto alloc_h = [&](size_t elems) -> _Float16* {   // f16
    off = (off + 255) & ~(size_t)255;
    _Float16* p = (_Float16*)(base + off);
    off += elems * 2;
    return p;
  };
  auto alloc_f = [&](size_t elems) -> float* {      // f32
    off = (off + 255) & ~(size_t)255;
    float* p = (float*)(base + off);
    off += elems * 4;
    return p;
  };
  auto hbuf = [&](int H, int Cp) { return alloc_h((size_t)(H + 2) * (H + 2) * Cp); };

  // activation buffers (NHWC f16, halo, padded channels)
  _Float16* A0 = hbuf(128, 32);                     // packed input (3 used)
  _Float16* A1 = hbuf(128, 96);                     // conv1 out (80 used)
  _Float16* A2 = hbuf(128, 96);                     // dbo(x)
  _Float16* A3 = hbuf(128, 160);                    // cell tmp
  _Float16* A4 = hbuf(128, 256);                    // cell tmp
  _Float16* A5 = hbuf(128, 160);                    // xr1
  _Float16* A6 = hbuf(128, 160);                    // xb1
  _Float16* A7 = hbuf(128, 160);                    // xr1+xb1
  _Float16* B0 = hbuf(64, 160);                     // blk1
  _Float16* B1 = hbuf(64, 160);                     // pooled xr1
  _Float16* B2 = hbuf(64, 160);                     // dbo
  _Float16* B3 = hbuf(64, 224);                     // MID tmp
  _Float16* B4 = hbuf(64, 224);                     // MID tmp
  _Float16* B5 = hbuf(64, 160);                     // xr2
  _Float16* B6 = hbuf(64, 160);                     // xb2
  _Float16* B7 = hbuf(64, 160);                     // sum
  _Float16* C0 = hbuf(32, 160);                     // blk2
  _Float16* C1 = hbuf(32, 160);                     // pooled xr2
  _Float16* C2 = hbuf(32, 160);                     // dbo
  _Float16* C3 = hbuf(32, 192);                     // HIGH tmp
  _Float16* C4 = hbuf(32, 192);                     // HIGH tmp
  _Float16* C5 = hbuf(32, 160);                     // xr3
  _Float16* C6 = hbuf(32, 160);                     // xb3
  _Float16* C7 = hbuf(32, 160);                     // sum
  _Float16* D0 = hbuf(16, 160);                     // blk3
  _Float16* Fc = hbuf(32, 480);                     // concat
  _Float16* Gb = hbuf(32, 160);                     // last1 out
  float* SPa = alloc_f(4096);                       // avg plane
  float* SPm = alloc_f(4096);                       // max plane
  float* SAt = alloc_f(4096);                       // attention plane

  // transformed weights: f16 [co][9][cip]
  struct CS { int ci, cip, co; };
  const CS low[3]  = {{80, 96, 160}, {160, 160, 256}, {256, 256, 160}};
  const CS mid[4]  = {{160, 160, 192}, {192, 224, 160}, {160, 224, 224}, {224, 224, 160}};
  const CS high[3] = {{160, 160, 160}, {160, 192, 192}, {192, 192, 160}};
  _Float16* Wc1 = alloc_h((size_t)80 * 9 * 32);
  _Float16 *Wb1[3], *Wb1d[3], *Wb2[4], *Wb2d[4], *Wb3[3], *Wb3d[3];
  for (int l = 0; l < 3; ++l) { Wb1[l]  = alloc_h((size_t)low[l].co * 9 * low[l].cip);
                                Wb1d[l] = alloc_h((size_t)low[l].co * 9 * low[l].cip); }
  for (int l = 0; l < 4; ++l) { Wb2[l]  = alloc_h((size_t)mid[l].co * 9 * mid[l].cip);
                                Wb2d[l] = alloc_h((size_t)mid[l].co * 9 * mid[l].cip); }
  for (int l = 0; l < 3; ++l) { Wb3[l]  = alloc_h((size_t)high[l].co * 9 * high[l].cip);
                                Wb3d[l] = alloc_h((size_t)high[l].co * 9 * high[l].cip); }
  _Float16* Wl1 = alloc_h((size_t)160 * 9 * 480);
  if (off > ws_size) return;                        // scratch too small; bail

  // zero halos + weight pads (and everything else) once per launch
  hipMemsetAsync(d_ws, 0, off, stream);

  // --- weight transforms (fold CDC, cast f16, [co][k3][ci_pad]) ---
  auto wt = [&](int g, int l, CS s, _Float16* dst) {
    int n = s.co * s.ci;
    wtrans<<<(n + 255) / 256, 256, 0, stream>>>(Pf(g, 3 * l), dst, s.co, s.ci, s.cip);
  };
  wt(G_CONV1, 0, CS{3, 32, 80}, Wc1);
  for (int l = 0; l < 3; ++l) { wt(G_B1, l, low[l], Wb1[l]); wt(G_B1D, l, low[l], Wb1d[l]); }
  for (int l = 0; l < 4; ++l) { wt(G_B2, l, mid[l], Wb2[l]); wt(G_B2D, l, mid[l], Wb2d[l]); }
  for (int l = 0; l < 3; ++l) { wt(G_B3, l, high[l], Wb3[l]); wt(G_B3D, l, high[l], Wb3d[l]); }
  wt(G_LAST1, 0, CS{480, 480, 160}, Wl1);

  auto conv = [&](const _Float16* in, const _Float16* w, const float* s, const float* b,
                  _Float16* o, int H, int Cinp, int Cout, int Coutp) {
    int tiles = (Cout / 16) * ((H * H) / 16);
    conv3_wmma<<<(tiles + 7) / 8, 256, 0, stream>>>(in, w, s, b, o, H, H,
                                                    Cinp, Cout, Coutp, tiles);
  };
  auto mpool = [&](const _Float16* in, _Float16* o, int Hi, int C, int Cpi, int Cpo) {
    int n = (Hi / 2) * (Hi / 2) * C;
    maxpool_kernel<<<(n + 255) / 256, 256, 0, stream>>>(in, o, Hi, C, Cpi, Cpo);
  };

  // --- stage 0: input pack + conv1 ---
  pack_x<<<(3 * 128 * 128 + 255) / 256, 256, 0, stream>>>(x, A0);
  conv(A0, Wc1, Pf(G_CONV1, 1), Pf(G_CONV1, 2), A1, 128, 32, 80, 96);

  // --- stage 1 (128x128, LOW cells) ---
  dbo_kernel<<<128 * 128, 256, 0, stream>>>(A1, A2, 128, 80, 96);
  conv(A1, Wb1[0], Pf(G_B1, 1), Pf(G_B1, 2), A3, 128, 96, 160, 160);
  conv(A3, Wb1[1], Pf(G_B1, 4), Pf(G_B1, 5), A4, 128, 160, 256, 256);
  conv(A4, Wb1[2], Pf(G_B1, 7), Pf(G_B1, 8), A5, 128, 256, 160, 160);
  conv(A2, Wb1d[0], Pf(G_B1D, 1), Pf(G_B1D, 2), A3, 128, 96, 160, 160);
  conv(A3, Wb1d[1], Pf(G_B1D, 4), Pf(G_B1D, 5), A4, 128, 160, 256, 256);
  conv(A4, Wb1d[2], Pf(G_B1D, 7), Pf(G_B1D, 8), A6, 128, 256, 160, 160);
  { int n = 130 * 130 * 160;
    add_kernel<<<(n + 255) / 256, 256, 0, stream>>>(A5, A6, A7, n); }
  mpool(A7, B0, 128, 160, 160, 160);   // blk1
  mpool(A5, B1, 128, 160, 160, 160);   // xr1 pooled

  // --- stage 2 (64x64, MID cells) ---
  dbo_kernel<<<64 * 64, 256, 0, stream>>>(B1, B2, 64, 160, 160);
  conv(B1, Wb2[0], Pf(G_B2, 1),  Pf(G_B2, 2),  B3, 64, 160, 192, 224);
  conv(B3, Wb2[1], Pf(G_B2, 4),  Pf(G_B2, 5),  B4, 64, 224, 160, 224);
  conv(B4, Wb2[2], Pf(G_B2, 7),  Pf(G_B2, 8),  B3, 64, 224, 224, 224);
  conv(B3, Wb2[3], Pf(G_B2, 10), Pf(G_B2, 11), B5, 64, 224, 160, 160);
  conv(B2, Wb2d[0], Pf(G_B2D, 1),  Pf(G_B2D, 2),  B3, 64, 160, 192, 224);
  conv(B3, Wb2d[1], Pf(G_B2D, 4),  Pf(G_B2D, 5),  B4, 64, 224, 160, 224);
  conv(B4, Wb2d[2], Pf(G_B2D, 7),  Pf(G_B2D, 8),  B3, 64, 224, 224, 224);
  conv(B3, Wb2d[3], Pf(G_B2D, 10), Pf(G_B2D, 11), B6, 64, 224, 160, 160);
  { int n = 66 * 66 * 160;
    add_kernel<<<(n + 255) / 256, 256, 0, stream>>>(B5, B6, B7, n); }
  mpool(B7, C0, 64, 160, 160, 160);    // blk2
  mpool(B5, C1, 64, 160, 160, 160);    // xr2 pooled

  // --- stage 3 (32x32, HIGH cells) ---
  dbo_kernel<<<32 * 32, 256, 0, stream>>>(C1, C2, 32, 160, 160);
  conv(C1, Wb3[0], Pf(G_B3, 1), Pf(G_B3, 2), C3, 32, 160, 160, 192);
  conv(C3, Wb3[1], Pf(G_B3, 4), Pf(G_B3, 5), C4, 32, 192, 192, 192);
  conv(C4, Wb3[2], Pf(G_B3, 7), Pf(G_B3, 8), C5, 32, 192, 160, 160);
  conv(C2, Wb3d[0], Pf(G_B3D, 1), Pf(G_B3D, 2), C3, 32, 160, 160, 192);
  conv(C3, Wb3d[1], Pf(G_B3D, 4), Pf(G_B3D, 5), C4, 32, 192, 192, 192);
  conv(C4, Wb3d[2], Pf(G_B3D, 7), Pf(G_B3D, 8), C6, 32, 192, 160, 160);
  { int n = 34 * 34 * 160;
    add_kernel<<<(n + 255) / 256, 256, 0, stream>>>(C5, C6, C7, n); }
  mpool(C7, D0, 32, 160, 160, 160);    // blk3 (16x16)

  // --- spatial attention + resize into concat buffer ---
  { int n = 32 * 32 * 160;
    sa_reduce<<<(64 * 64 + 255) / 256, 256, 0, stream>>>(B0, SPa, SPm, 64, 160, 160);
    sa_conv<<<(64 * 64 + 255) / 256, 256, 0, stream>>>(SPa, SPm, Pf(G_SA1, 0), 64, 7, SAt);
    attn_down2<<<(n + 255) / 256, 256, 0, stream>>>(B0, SAt, Fc, 64, 160, 160, 0);
    sa_reduce<<<(32 * 32 + 255) / 256, 256, 0, stream>>>(C0, SPa, SPm, 32, 160, 160);
    sa_conv<<<(32 * 32 + 255) / 256, 256, 0, stream>>>(SPa, SPm, Pf(G_SA2, 0), 32, 5, SAt);
    attn_ident<<<(n + 255) / 256, 256, 0, stream>>>(C0, SAt, Fc, 160, 160, 160);
    sa_reduce<<<(16 * 16 + 255) / 256, 256, 0, stream>>>(D0, SPa, SPm, 16, 160, 160);
    sa_conv<<<(16 * 16 + 255) / 256, 256, 0, stream>>>(SPa, SPm, Pf(G_SA3, 0), 16, 3, SAt);
    attn_up2<<<(n + 255) / 256, 256, 0, stream>>>(D0, SAt, Fc, 16, 160, 160, 320);
  }

  // --- head: last1 (480->160, WMMA) + last2 (160->1, f32 out) ---
  conv(Fc, Wl1, Pf(G_LAST1, 1), Pf(G_LAST1, 2), Gb, 32, 480, 160, 160);
  last2_kernel<<<4, 256, 0, stream>>>(Gb, Pf(G_LAST2, 0), (float*)d_out);
}
